// DeformableTransformerEncoder_43748536877317
// MI455X (gfx1250) — matrix operand order repaired and checked by reading, not compiled, not run on hardware
//
#include <hip/hip_runtime.h>

// ---------------- problem constants (from reference) ----------------
#define S_TOT   5440
#define C_DIM   256
#define F_DIM   1024
#define NLAYERS 6
#define NHEADS  8

typedef __attribute__((ext_vector_type(16))) _Float16 v16h;
typedef __attribute__((ext_vector_type(8)))  _Float16 half8;
typedef __attribute__((ext_vector_type(8)))  float    v8f;

union FragAB { v16h v; half8 h[2]; };

__device__ __forceinline__ float wave_reduce_sum(float v) {
#pragma unroll
  for (int off = 16; off > 0; off >>= 1) v += __shfl_xor(v, off, 32);
  return v;
}

// ---------------- WMMA GEMM: C[M,N] = A[M,K](f16) * Bt[N,K](f16)^T + bias ----------------
// Wave tile: 32(M) x 64(N) -> 2 A-frags x 4 B-frags -> 8 WMMAs per k-step (K += 32).
// EPI==0: f32 output + bias.  EPI==1: relu(..)+bias -> f16 output (FFN hidden).
template <int EPI>
__global__ void gemm16(const _Float16* __restrict__ A, const _Float16* __restrict__ Bt,
                       const float* __restrict__ bias, float* __restrict__ Cf,
                       _Float16* __restrict__ Ch, int M, int N, int K) {
  const int wave = threadIdx.x >> 5;
  const int lane = threadIdx.x & 31;
  const int gid  = blockIdx.x * 4 + wave;
  const int tN   = N >> 6;                  // 64-wide N tiles
  const int tiles = (M >> 5) * tN;          // 32-tall M tiles
  if (gid >= tiles) return;                 // wave-uniform: EXEC stays all-ones for WMMA
  const int tm = gid / tN, tn = gid % tN;
  const int h = lane >> 4, r = lane & 15;

  const _Float16* __restrict__ arow0 = A + (size_t)(tm * 32 + r) * K;
  const _Float16* __restrict__ arow1 = A + (size_t)(tm * 32 + 16 + r) * K;
  const _Float16* __restrict__ brow[4] = {
      Bt + (size_t)(tn * 64 + 0 * 16 + r) * K,
      Bt + (size_t)(tn * 64 + 1 * 16 + r) * K,
      Bt + (size_t)(tn * 64 + 2 * 16 + r) * K,
      Bt + (size_t)(tn * 64 + 3 * 16 + r) * K};

  v8f c[2][4];
#pragma unroll
  for (int i = 0; i < 2; ++i)
#pragma unroll
    for (int j = 0; j < 4; ++j) c[i][j] = v8f{0.f, 0.f, 0.f, 0.f, 0.f, 0.f, 0.f, 0.f};

  for (int k0 = 0; k0 < K; k0 += 32) {
    FragAB a[2], b[4];
    // A frag: VGPR0-3 -> K = k0+8h .. +7 ; VGPR4-7 -> K = k0+16+8h .. +7
    a[0].h[0] = *(const half8*)(arow0 + k0 + h * 8);
    a[0].h[1] = *(const half8*)(arow0 + k0 + 16 + h * 8);
    a[1].h[0] = *(const half8*)(arow1 + k0 + h * 8);
    a[1].h[1] = *(const half8*)(arow1 + k0 + 16 + h * 8);
    // B frag: VGPR0-3 -> K = k0+8h .. +7 ; VGPR4-7 -> K = k0+16+8h .. +7
#pragma unroll
    for (int j = 0; j < 4; ++j) {
      b[j].h[0] = *(const half8*)(brow[j] + k0 + h * 8);
      b[j].h[1] = *(const half8*)(brow[j] + k0 + 16 + h * 8);
    }
#pragma unroll
    for (int i = 0; i < 2; ++i)
#pragma unroll
      for (int j = 0; j < 4; ++j)
        c[i][j] = __builtin_amdgcn_wmma_f32_16x16x32_f16(false, a[i].v, false, b[j].v,
                                                         (short)0, c[i][j], false, false);
  }

#pragma unroll
  for (int j = 0; j < 4; ++j) {
    const int n = tn * 64 + j * 16 + r;
    const float bn = bias[n];
#pragma unroll
    for (int i = 0; i < 2; ++i) {
#pragma unroll
      for (int rr = 0; rr < 8; ++rr) {
        const int m = tm * 32 + i * 16 + h * 8 + rr;   // C layout: VGPR rr -> row rr + 8h
        float val = c[i][j][rr] + bn;
        if (EPI == 1) {
          val = val > 0.f ? val : 0.f;
          Ch[(size_t)m * N + n] = (_Float16)val;
        } else {
          Cf[(size_t)m * N + n] = val;
        }
      }
    }
  }
}

// ---------------- weight transpose + f32->f16 convert: Wt[n*K+k] = W[k*N+n] ----------------
__global__ void wprep(const float* __restrict__ W, _Float16* __restrict__ Wt, int K, int N) {
  int t = blockIdx.x * blockDim.x + threadIdx.x;
  if (t >= K * N) return;
  int nn = t / K, kk = t % K;
  Wt[t] = (_Float16)W[(size_t)kk * N + nn];
}

// ---------------- x/x16/q16 prep: x = xs; x16 = f16(xs); q16 = f16(xs + pos) ----------------
__global__ void prep_inputs(const float* __restrict__ xs, const float* __restrict__ pos,
                            float* __restrict__ x, _Float16* __restrict__ x16,
                            _Float16* __restrict__ q16, int n) {
  int t = blockIdx.x * blockDim.x + threadIdx.x;
  if (t >= n) return;
  float xv = xs[t];
  x[t] = xv;
  x16[t] = (_Float16)xv;
  q16[t] = (_Float16)(xv + pos[t]);
}

// ---------------- softmax over 16 contiguous logits per (s,head), in place ----------------
__global__ void softmax16(float* __restrict__ logits, int groups) {
  int t = blockIdx.x * blockDim.x + threadIdx.x;
  if (t >= groups) return;
  float* p = logits + (size_t)t * 16;
  float e[16];
  float mx = p[0];
#pragma unroll
  for (int i = 1; i < 16; ++i) mx = fmaxf(mx, p[i]);
  float sum = 0.f;
#pragma unroll
  for (int i = 0; i < 16; ++i) { e[i] = __expf(p[i] - mx); sum += e[i]; }
  float inv = 1.f / sum;
#pragma unroll
  for (int i = 0; i < 16; ++i) p[i] = e[i] * inv;
}

// ---------------- MSDA bilinear sampling core: one wave per (s, head), lane = channel ----------------
__global__ void msda_sample(const float* __restrict__ value, const float* __restrict__ off,
                            const float* __restrict__ aw, _Float16* __restrict__ out) {
  const int wave = threadIdx.x >> 5, lane = threadIdx.x & 31;
  const int gid = blockIdx.x * 4 + wave;
  if (gid >= S_TOT * NHEADS) return;
  const int s = gid >> 3, hh = gid & 7;

  const int HW[4] = {64, 32, 16, 8};          // square levels
  const int ST[4] = {0, 4096, 5120, 5376};

  // reference point of query s (valid_ratios == 1)
  int lvl = (s < 4096) ? 0 : (s < 5120 ? 1 : (s < 5376 ? 2 : 3));
  int Wq = HW[lvl];
  int idx = s - ST[lvl];
  int row = idx / Wq, col = idx % Wq;
  float refx = (col + 0.5f) / (float)Wq;
  float refy = (row + 0.5f) / (float)Wq;

  const float* __restrict__ offp = off + (size_t)s * 256 + hh * 32;   // (h,l,p,2)
  const float* __restrict__ awp  = aw  + (size_t)s * 128 + hh * 16;   // (h,l,p)

  float acc = 0.f;
#pragma unroll
  for (int l = 0; l < 4; ++l) {
    const int Wl = HW[l], Hl = HW[l], s0 = ST[l];
#pragma unroll
    for (int p = 0; p < 4; ++p) {
      float ox = offp[l * 8 + p * 2 + 0];
      float oy = offp[l * 8 + p * 2 + 1];
      float x = (refx + ox / (float)Wl) * (float)Wl - 0.5f;
      float y = (refy + oy / (float)Hl) * (float)Hl - 0.5f;
      float x0f = floorf(x), y0f = floorf(y);
      float lx = x - x0f, ly = y - y0f;
      int x0 = (int)x0f, y0 = (int)y0f;
      float w = awp[l * 4 + p];
      float v00 = 0.f, v01 = 0.f, v10 = 0.f, v11 = 0.f;
      bool xi0 = (x0 >= 0) & (x0 < Wl);
      bool xi1 = (x0 + 1 >= 0) & (x0 + 1 < Wl);
      if (y0 >= 0 && y0 < Hl) {
        if (xi0) v00 = value[(size_t)(s0 + y0 * Wl + x0) * 256 + hh * 32 + lane];
        if (xi1) v01 = value[(size_t)(s0 + y0 * Wl + x0 + 1) * 256 + hh * 32 + lane];
      }
      if (y0 + 1 >= 0 && y0 + 1 < Hl) {
        if (xi0) v10 = value[(size_t)(s0 + (y0 + 1) * Wl + x0) * 256 + hh * 32 + lane];
        if (xi1) v11 = value[(size_t)(s0 + (y0 + 1) * Wl + x0 + 1) * 256 + hh * 32 + lane];
      }
      float samp = v00 * (1.f - lx) * (1.f - ly) + v01 * lx * (1.f - ly)
                 + v10 * (1.f - lx) * ly         + v11 * lx * ly;
      acc += w * samp;
    }
  }
  out[(size_t)s * 256 + hh * 32 + lane] = (_Float16)acc;
}

// ---------------- residual + LayerNorm: x = LN(x + a); also writes f16 copy ----------------
__global__ void ln_residual(const float* __restrict__ x, const float* __restrict__ a,
                            const float* __restrict__ g, const float* __restrict__ b,
                            float* __restrict__ xo, _Float16* __restrict__ xo16) {
  const int wave = threadIdx.x >> 5, lane = threadIdx.x & 31;
  const int rowI = blockIdx.x * 4 + wave;
  if (rowI >= S_TOT) return;
  const float* xr = x + (size_t)rowI * C_DIM;
  const float* ar = a + (size_t)rowI * C_DIM;
  float v[8];
  float s = 0.f;
#pragma unroll
  for (int i = 0; i < 8; ++i) { v[i] = xr[lane + i * 32] + ar[lane + i * 32]; s += v[i]; }
  s = wave_reduce_sum(s);
  float mean = s * (1.f / 256.f);
  float sq = 0.f;
#pragma unroll
  for (int i = 0; i < 8; ++i) { float d = v[i] - mean; sq += d * d; }
  sq = wave_reduce_sum(sq);
  float rstd = rsqrtf(sq * (1.f / 256.f) + 1e-5f);
#pragma unroll
  for (int i = 0; i < 8; ++i) {
    int cidx = lane + i * 32;
    float o = (v[i] - mean) * rstd * g[cidx] + b[cidx];
    xo[(size_t)rowI * C_DIM + cidx] = o;
    xo16[(size_t)rowI * C_DIM + cidx] = (_Float16)o;
  }
}

// ---------------- host orchestration ----------------
extern "C" void kernel_launch(void* const* d_in, const int* in_sizes, int n_in,
                              void* d_out, int out_size, void* d_ws, size_t ws_size,
                              hipStream_t stream) {
  (void)in_sizes; (void)n_in; (void)out_size; (void)ws_size;

  const float* src   = (const float*)d_in[0];
  const float* pos   = (const float*)d_in[1];
  const float* Wv    = (const float*)d_in[5];
  const float* bv    = (const float*)d_in[6];
  const float* Woff  = (const float*)d_in[7];
  const float* boff  = (const float*)d_in[8];
  const float* Wattn = (const float*)d_in[9];
  const float* battn = (const float*)d_in[10];
  const float* Wout  = (const float*)d_in[11];
  const float* bout  = (const float*)d_in[12];
  const float* ln1g  = (const float*)d_in[13];
  const float* ln1b  = (const float*)d_in[14];
  const float* W1    = (const float*)d_in[15];
  const float* b1    = (const float*)d_in[16];
  const float* W2    = (const float*)d_in[17];
  const float* b2    = (const float*)d_in[18];
  const float* ln2g  = (const float*)d_in[19];
  const float* ln2b  = (const float*)d_in[20];

  const size_t SC = (size_t)S_TOT * C_DIM;          // 1,392,640
  char* ws = (char*)d_ws;
  size_t o = 0;
  float*    x    = (float*)(ws + o);    o += SC * 4;
  _Float16* x16  = (_Float16*)(ws + o); o += SC * 2;
  _Float16* q16  = (_Float16*)(ws + o); o += SC * 2;
  float*    val  = (float*)(ws + o);    o += SC * 4;
  float*    offb = (float*)(ws + o);    o += SC * 4;
  float*    attn = (float*)(ws + o);    o += (size_t)S_TOT * 128 * 4;
  _Float16* m16  = (_Float16*)(ws + o); o += SC * 2;
  float*    gout = (float*)(ws + o);    o += SC * 4;
  _Float16* h16  = (_Float16*)(ws + o); o += (size_t)S_TOT * F_DIM * 2;
  _Float16* wt   = (_Float16*)(ws + o); o += 753664ull * 2;             // per-layer f16^T weights

  // per-layer transposed-weight sub-buffers
  _Float16* wtV = wt;                // 256x256  -> 65536
  _Float16* wtF = wt + 65536;        // 256x256
  _Float16* wtA = wt + 131072;       // 128x256  -> 32768
  _Float16* wtO = wt + 163840;       // 256x256
  _Float16* wt1 = wt + 229376;       // 1024x256 -> 262144
  _Float16* wt2 = wt + 491520;       // 256x1024 -> 262144

  for (int i = 0; i < NLAYERS; ++i) {
    const float* Wv_i  = Wv  + (size_t)i * 65536;
    const float* Wf_i  = Woff + (size_t)i * 65536;
    const float* Wa_i  = Wattn + (size_t)i * 32768;
    const float* Wo_i  = Wout + (size_t)i * 65536;
    const float* W1_i  = W1 + (size_t)i * 262144;
    const float* W2_i  = W2 + (size_t)i * 262144;
    const float* bv_i  = bv + (size_t)i * 256;
    const float* bf_i  = boff + (size_t)i * 256;
    const float* ba_i  = battn + (size_t)i * 128;
    const float* bo_i  = bout + (size_t)i * 256;
    const float* b1_i  = b1 + (size_t)i * 1024;
    const float* b2_i  = b2 + (size_t)i * 256;

    // stage f16 transposed weights for this layer
    wprep<<<256, 256, 0, stream>>>(Wv_i, wtV, 256, 256);
    wprep<<<256, 256, 0, stream>>>(Wf_i, wtF, 256, 256);
    wprep<<<128, 256, 0, stream>>>(Wa_i, wtA, 256, 128);
    wprep<<<256, 256, 0, stream>>>(Wo_i, wtO, 256, 256);
    wprep<<<1024, 256, 0, stream>>>(W1_i, wt1, 256, 1024);
    wprep<<<1024, 256, 0, stream>>>(W2_i, wt2, 1024, 256);

    // x / x16 / q16 (layer 0 reads src; later layers read x in place)
    prep_inputs<<<(int)(SC / 256), 256, 0, stream>>>(i == 0 ? src : x, pos, x, x16, q16, (int)SC);

    // projections: wave-tiles = (M/32)*(N/64); 4 waves per block
    gemm16<0><<<170, 128, 0, stream>>>(x16, wtV, bv_i, val,  nullptr, S_TOT, 256, 256);
    gemm16<0><<<170, 128, 0, stream>>>(q16, wtF, bf_i, offb, nullptr, S_TOT, 256, 256);
    gemm16<0><<<85,  128, 0, stream>>>(q16, wtA, ba_i, attn, nullptr, S_TOT, 128, 256);

    softmax16<<<170, 256, 0, stream>>>(attn, S_TOT * NHEADS);
    msda_sample<<<10880, 128, 0, stream>>>(val, offb, attn, m16);

    // output projection + residual LN1
    gemm16<0><<<170, 128, 0, stream>>>(m16, wtO, bo_i, gout, nullptr, S_TOT, 256, 256);
    ln_residual<<<1360, 128, 0, stream>>>(x, gout, ln1g + (size_t)i * 256, ln1b + (size_t)i * 256, x, x16);

    // FFN + residual LN2
    gemm16<1><<<680, 128, 0, stream>>>(x16, wt1, b1_i, nullptr, h16, S_TOT, 1024, 256);
    gemm16<0><<<170, 128, 0, stream>>>(h16, wt2, b2_i, gout, nullptr, S_TOT, 256, 1024);
    ln_residual<<<1360, 128, 0, stream>>>(x, gout, ln2g + (size_t)i * 256, ln2b + (size_t)i * 256, x, x16);
  }

  hipMemcpyAsync(d_out, x, SC * sizeof(float), hipMemcpyDeviceToDevice, stream);
}
